// MultiHeadAttentionLayer_79680233276100
// MI455X (gfx1250) — matrix operand order repaired and checked by reading, not compiled, not run on hardware
//
#include <hip/hip_runtime.h>
#include <math.h>

typedef __attribute__((ext_vector_type(16))) __bf16 v16bf;
typedef __attribute__((ext_vector_type(8)))  float  v8f;

#define DMODEL 1024
#define NHEAD  16
#define HDIM   64
#define SEQ    2048
#define BATCH  4
#define MROWS  (BATCH*SEQ)

// ---------------------------------------------------------------- helpers
static __device__ __forceinline__ unsigned short f2bf(float f) {
  __bf16 h = (__bf16)f;
  unsigned short u;
  __builtin_memcpy(&u, &h, 2);
  return u;
}

// A-fragment (16x32 bf16, M x K): lane<16 holds K={0..7,16..23}, lane>=16 K={8..15,24..31}
// caller passes p = rowbase + (lane>>4)*8 ; chunks at +0 and +16 halfs
static __device__ __forceinline__ v16bf load_frag_a(const unsigned short* p) {
  union { v16bf v; uint4 q[2]; } u;
  u.q[0] = *(const uint4*)(p);
  u.q[1] = *(const uint4*)(p + 16);
  return u.v;
}
// B-fragment (32x16 bf16, K x N) loaded from B^T rows [N][K]:
// lane<16 holds K={0..7,16..23}, lane>=16 K={8..15,24..31}
// caller passes p = rowbase + (lane>>4)*8 ; chunks at +0 and +16 halfs
static __device__ __forceinline__ v16bf load_frag_b(const unsigned short* p) {
  union { v16bf v; uint4 q[2]; } u;
  u.q[0] = *(const uint4*)(p);
  u.q[1] = *(const uint4*)(p + 16);
  return u.v;
}

static __device__ __forceinline__ float rmax16(float x) {
  x = fmaxf(x, __shfl_xor(x, 1));
  x = fmaxf(x, __shfl_xor(x, 2));
  x = fmaxf(x, __shfl_xor(x, 4));
  x = fmaxf(x, __shfl_xor(x, 8));
  return x;
}
static __device__ __forceinline__ float rsum16(float x) {
  x += __shfl_xor(x, 1);
  x += __shfl_xor(x, 2);
  x += __shfl_xor(x, 4);
  x += __shfl_xor(x, 8);
  return x;
}

// gfx1250 async copy: global memory -> LDS, 16 bytes per lane, ASYNCcnt-tracked.
// lds_off = LDS byte offset (low 32 bits of generic pointer to __shared__).
static __device__ __forceinline__ void async_load_b128(unsigned lds_off,
                                                       const void* gptr) {
  asm volatile("global_load_async_to_lds_b128 %0, %1, off"
               :
               : "v"(lds_off), "v"(gptr)
               : "memory");
}
static __device__ __forceinline__ void wait_async0() {
  asm volatile("s_wait_asynccnt 0" ::: "memory");
}

// ---------------------------------------------------------------- GEMM
// C[M,N] = A[M,K] * W[K,N] + bias ;  M=8192, N=K=1024
// MODE 0: write bf16 to [B,H,S,Dh]   (Q, K projections)
// MODE 1: write bf16 to [B,H,Dh,S]   (V projection, pre-transposed)
// MODE 2: write f32  to [M,N]        (output projection)
enum { MODE_BHSD = 0, MODE_BHDS = 1, MODE_F32 = 2 };

template <int MODE>
__global__ __launch_bounds__(256) void gemm_bias(
    const float* __restrict__ A, const float* __restrict__ W,
    const float* __restrict__ bias, void* __restrict__ out, float out_scale) {
  __shared__ alignas(16) unsigned short lds_a[2][128 * 40];  // [row][k], pad 40
  __shared__ alignas(16) unsigned short lds_b[2][128 * 40];  // [n][k],   pad 40

  const int tid  = threadIdx.x;
  const int lane = tid & 31;
  const int w    = tid >> 5;   // 0..7
  const int wm   = w >> 1;     // 0..3 : 32-row group
  const int wn   = w & 1;      // 0..1 : 64-col group
  const int ln   = lane & 15;
  const int lh   = lane >> 4;
  const long rowBlk = (long)blockIdx.x * 128;
  const int  colBlk = blockIdx.y * 128;

  v8f acc[2][4];
#pragma unroll
  for (int i = 0; i < 2; i++)
#pragma unroll
    for (int j = 0; j < 4; j++) acc[i][j] = (v8f){};

  float4 ra[4], rw[4];
  auto load_tiles = [&](int k0) {
#pragma unroll
    for (int it = 0; it < 4; it++) {
      int idx = tid + it * 256;  // 0..1023 float4 chunks
      ra[it]  = *(const float4*)(A + (rowBlk + (idx >> 3)) * DMODEL + k0 +
                                ((idx & 7) << 2));
      rw[it]  = *(const float4*)(W + (long)(k0 + (idx >> 5)) * DMODEL + colBlk +
                                ((idx & 31) << 2));
    }
  };
  auto store_tiles = [&](int buf) {
#pragma unroll
    for (int it = 0; it < 4; it++) {
      int idx = tid + it * 256;
      int ar  = idx >> 3;             // A row 0..127
      int ac  = (idx & 7) << 2;       // A k
      uint2 pk;
      pk.x = (unsigned)f2bf(ra[it].x) | ((unsigned)f2bf(ra[it].y) << 16);
      pk.y = (unsigned)f2bf(ra[it].z) | ((unsigned)f2bf(ra[it].w) << 16);
      *(uint2*)(lds_a[buf] + ar * 40 + ac) = pk;
      int kr = idx >> 5;              // W k 0..31
      int nc = (idx & 31) << 2;       // W col
      lds_b[buf][(nc + 0) * 40 + kr] = f2bf(rw[it].x);
      lds_b[buf][(nc + 1) * 40 + kr] = f2bf(rw[it].y);
      lds_b[buf][(nc + 2) * 40 + kr] = f2bf(rw[it].z);
      lds_b[buf][(nc + 3) * 40 + kr] = f2bf(rw[it].w);
    }
  };

  load_tiles(0);
  store_tiles(0);
  __syncthreads();

  for (int k = 0; k < DMODEL / 32; k++) {
    const int buf = k & 1;
    if (k + 1 < DMODEL / 32) load_tiles((k + 1) * 32);

    v16bf af[2], bfr[4];
#pragma unroll
    for (int tm = 0; tm < 2; tm++)
      af[tm] = load_frag_a(lds_a[buf] + (wm * 32 + tm * 16 + ln) * 40 + lh * 8);
#pragma unroll
    for (int tn = 0; tn < 4; tn++)
      bfr[tn] = load_frag_b(lds_b[buf] + (wn * 64 + tn * 16 + ln) * 40 + lh * 8);
#pragma unroll
    for (int tm = 0; tm < 2; tm++)
#pragma unroll
      for (int tn = 0; tn < 4; tn++)
        acc[tm][tn] = __builtin_amdgcn_wmma_f32_16x16x32_bf16(
            false, af[tm], false, bfr[tn], (short)0, acc[tm][tn], false, false);

    if (k + 1 < DMODEL / 32) store_tiles(buf ^ 1);  // other buffer: no race
    __syncthreads();
  }

  // epilogue
#pragma unroll
  for (int tm = 0; tm < 2; tm++) {
#pragma unroll
    for (int tn = 0; tn < 4; tn++) {
      int gcol  = colBlk + wn * 64 + tn * 16 + ln;
      float bvs = bias[gcol];
#pragma unroll
      for (int r = 0; r < 8; r++) {
        long grow = rowBlk + wm * 32 + tm * 16 + r + lh * 8;
        float val = (acc[tm][tn][r] + bvs) * out_scale;
        if (MODE == MODE_F32) {
          ((float*)out)[grow * DMODEL + gcol] = val;
        } else {
          int b  = (int)(grow >> 11);      // / 2048
          int s  = (int)(grow & 2047);
          int hh = gcol >> 6;
          int dh = gcol & 63;
          long addr;
          if (MODE == MODE_BHSD)
            addr = (((long)(b * NHEAD + hh)) * SEQ + s) * HDIM + dh;
          else
            addr = (((long)(b * NHEAD + hh)) * HDIM + dh) * SEQ + s;
          ((unsigned short*)out)[addr] = f2bf(val);
        }
      }
    }
  }
}

// ---------------------------------------------------------------- attention
// Q pre-scaled by 1/sqrt(Dh). Each wave: 16 query rows; key tiles of 64.
__global__ __launch_bounds__(256) void flash_attn(
    const unsigned short* __restrict__ Qb,   // [B,H,S,Dh] bf16
    const unsigned short* __restrict__ Kb,   // [B,H,S,Dh] bf16
    const unsigned short* __restrict__ Vt,   // [B,H,Dh,S] bf16
    const int* __restrict__ mask,            // [B,1,1,S]
    float* __restrict__ ctx) {               // [B,S,D] fp32
  __shared__ alignas(16) unsigned short lds_k[64 * 72];       // [key][d]
  __shared__ alignas(16) unsigned short lds_v[64 * 72];       // [d][key]
  __shared__ alignas(16) unsigned short lds_p[8 * 16 * 72];   // per-wave P

  const int tid  = threadIdx.x;
  const int lane = tid & 31, w = tid >> 5;
  const int ln   = lane & 15, lh = lane >> 4;
  const int b    = blockIdx.z, h = blockIdx.y;
  const int q0   = blockIdx.x * 128 + w * 16;
  const long bh  = (long)(b * NHEAD + h);

  // Q fragments for this wave's 16 rows (16 x 64 -> two 16x32 A-frags)
  v16bf aq[2];
  {
    const unsigned short* qp = Qb + (bh * SEQ + q0 + ln) * HDIM;
    aq[0] = load_frag_a(qp + lh * 8);
    aq[1] = load_frag_a(qp + 32 + lh * 8);
  }

  v8f o[4];
#pragma unroll
  for (int j = 0; j < 4; j++) o[j] = (v8f){};
  float mrun[8], lrun[8];
#pragma unroll
  for (int r = 0; r < 8; r++) { mrun[r] = -__builtin_inff(); lrun[r] = 0.f; }
  unsigned short* pw = lds_p + w * 16 * 72;
  const float LOG2E = 1.44269504088896f;

  for (int kt = 0; kt < SEQ / 64; kt++) {
    // stage K tile [64 keys][64 d] and V^T tile [64 d][64 keys] via async DMA
#pragma unroll
    for (int it = 0; it < 2; it++) {
      int idx = tid + it * 256;          // 0..511 : 8-half chunks
      int r   = idx >> 3;
      int c8  = (idx & 7) << 3;
      async_load_b128((unsigned)(size_t)(lds_k + r * 72 + c8),
                      Kb + (bh * SEQ + kt * 64 + r) * HDIM + c8);
      async_load_b128((unsigned)(size_t)(lds_v + r * 72 + c8),
                      Vt + (bh * HDIM + r) * SEQ + kt * 64 + c8);
    }
    wait_async0();
    __syncthreads();

    // scores: 16 x 64 (4 tiles of 16x16), K-dim = Dh = 64 (2 wmma each)
    v8f s[4];
#pragma unroll
    for (int j = 0; j < 4; j++) {
      v8f z = (v8f){};
      v16bf bk0 = load_frag_b(lds_k + (j * 16 + ln) * 72 + lh * 8);
      v16bf bk1 = load_frag_b(lds_k + (j * 16 + ln) * 72 + 32 + lh * 8);
      z = __builtin_amdgcn_wmma_f32_16x16x32_bf16(false, aq[0], false, bk0,
                                                  (short)0, z, false, false);
      z = __builtin_amdgcn_wmma_f32_16x16x32_bf16(false, aq[1], false, bk1,
                                                  (short)0, z, false, false);
      int col    = kt * 64 + j * 16 + ln;   // lane = score column = key index
      float madd = (mask[b * SEQ + col] == 0) ? -1e9f : 0.f;
#pragma unroll
      for (int r = 0; r < 8; r++) z[r] += madd;
      s[j] = z;
    }

    // online softmax (C layout: VGPR r = row, lane = col within 16-group)
    float mnew[8], alpha[8];
#pragma unroll
    for (int r = 0; r < 8; r++) {
      float mx = fmaxf(fmaxf(s[0][r], s[1][r]), fmaxf(s[2][r], s[3][r]));
      mx       = rmax16(mx);
      mnew[r]  = fmaxf(mrun[r], mx);
      alpha[r] = exp2f((mrun[r] - mnew[r]) * LOG2E);
      mrun[r]  = mnew[r];
    }
    float rsum[8];
#pragma unroll
    for (int r = 0; r < 8; r++) rsum[r] = 0.f;
#pragma unroll
    for (int j = 0; j < 4; j++)
#pragma unroll
      for (int r = 0; r < 8; r++) {
        float p = exp2f((s[j][r] - mnew[r]) * LOG2E);
        s[j][r] = p;
        rsum[r] += p;
      }
#pragma unroll
    for (int r = 0; r < 8; r++)
      lrun[r] = lrun[r] * alpha[r] + rsum16(rsum[r]);
#pragma unroll
    for (int j = 0; j < 4; j++)
#pragma unroll
      for (int r = 0; r < 8; r++) o[j][r] *= alpha[r];

    // P (C layout) -> bf16 LDS [16 rows][64 keys], per-wave region
#pragma unroll
    for (int j = 0; j < 4; j++)
#pragma unroll
      for (int r = 0; r < 8; r++)
        pw[(r + lh * 8) * 72 + j * 16 + ln] = f2bf(s[j][r]);

    // reload P as A-fragments (DS ops in-order within wave)
    v16bf ap0 = load_frag_a(pw + ln * 72 + lh * 8);
    v16bf ap1 = load_frag_a(pw + ln * 72 + 32 + lh * 8);

    // O += P * V : 4 output d-tiles, K-dim = 64 keys (2 wmma each)
#pragma unroll
    for (int jn = 0; jn < 4; jn++) {
      v16bf bv0 = load_frag_b(lds_v + (jn * 16 + ln) * 72 + lh * 8);
      v16bf bv1 = load_frag_b(lds_v + (jn * 16 + ln) * 72 + 32 + lh * 8);
      o[jn] = __builtin_amdgcn_wmma_f32_16x16x32_bf16(false, ap0, false, bv0,
                                                      (short)0, o[jn], false, false);
      o[jn] = __builtin_amdgcn_wmma_f32_16x16x32_bf16(false, ap1, false, bv1,
                                                      (short)0, o[jn], false, false);
    }
    __syncthreads();
  }

  // normalize and write ctx [B,S,D] fp32
  float inv[8];
#pragma unroll
  for (int r = 0; r < 8; r++) inv[r] = 1.f / lrun[r];
#pragma unroll
  for (int jn = 0; jn < 4; jn++)
#pragma unroll
    for (int r = 0; r < 8; r++) {
      long row = (long)b * SEQ + q0 + r + lh * 8;
      int  col = h * HDIM + jn * 16 + ln;
      ctx[row * DMODEL + col] = o[jn][r] * inv[r];
    }
}

// ---------------------------------------------------------------- launch
extern "C" void kernel_launch(void* const* d_in, const int* in_sizes, int n_in,
                              void* d_out, int out_size, void* d_ws, size_t ws_size,
                              hipStream_t stream) {
  (void)in_sizes; (void)n_in; (void)out_size; (void)ws_size;
  const float* query = (const float*)d_in[0];
  const float* key_  = (const float*)d_in[1];
  const float* value = (const float*)d_in[2];
  const int*   mask  = (const int*)d_in[3];
  const float* Wq = (const float*)d_in[4];  const float* bq = (const float*)d_in[5];
  const float* Wk = (const float*)d_in[6];  const float* bk = (const float*)d_in[7];
  const float* Wv = (const float*)d_in[8];  const float* bv = (const float*)d_in[9];
  const float* Wo = (const float*)d_in[10]; const float* bo = (const float*)d_in[11];

  char* ws = (char*)d_ws;
  const size_t sz16 = (size_t)MROWS * DMODEL * sizeof(unsigned short); // 16 MB
  unsigned short* Qb  = (unsigned short*)(ws);
  unsigned short* Kb  = (unsigned short*)(ws + sz16);
  unsigned short* Vtb = (unsigned short*)(ws + 2 * sz16);
  float*          ctx = (float*)(ws + 3 * sz16);                        // 32 MB

  dim3 gg(MROWS / 128, DMODEL / 128);
  // Q projection with 1/sqrt(Dh) folded in
  gemm_bias<MODE_BHSD><<<gg, 256, 0, stream>>>(query, Wq, bq, Qb, 0.125f);
  gemm_bias<MODE_BHSD><<<gg, 256, 0, stream>>>(key_,  Wk, bk, Kb, 1.0f);
  gemm_bias<MODE_BHDS><<<gg, 256, 0, stream>>>(value, Wv, bv, Vtb, 1.0f);
  flash_attn<<<dim3(SEQ / 128, NHEAD, BATCH), 256, 0, stream>>>(Qb, Kb, Vtb, mask, ctx);
  gemm_bias<MODE_F32><<<gg, 256, 0, stream>>>(ctx, Wo, bo, (float*)d_out, 1.0f);
}
